// CrossAttentionBlock_50766513439444
// MI455X (gfx1250) — matrix operand order repaired and checked by reading, compiled, not run on hardware
//
#include <hip/hip_runtime.h>
#include <hip/hip_bf16.h>

// ---------------------------------------------------------------------------
// CrossAttentionBlock for MI455X (gfx1250, wave32, WMMA 16x16x32 f16->f32,
// Tensor Data Mover for global->LDS staging)
//
//   x  : (256, 256, 64) f32   Wq/Wk/Wv : (8,256,32) f32
//   Wp : (256,256) f32        bp : (256,) f32
//   out: (256, 256, 64) f32
// ---------------------------------------------------------------------------

typedef __attribute__((ext_vector_type(16))) _Float16 v16h;
typedef __attribute__((ext_vector_type(8)))  float    v8f;
typedef __attribute__((ext_vector_type(4)))  unsigned int u32x4;
typedef __attribute__((ext_vector_type(8)))  int      i32x8;
typedef __attribute__((ext_vector_type(4)))  int      i32x4;

#define F_DIM   256
#define HEADS   8
#define D_DIM   32
#define NAG     8
#define NB      32
#define SEQ     64
#define NTOK    (NAG * NB * SEQ)          // 16384
#define WFRAG   (F_DIM * F_DIM)           // 65536 halves per packed weight

#if defined(__HIP_DEVICE_COMPILE__) && \
    __has_builtin(__builtin_amdgcn_tensor_load_to_lds) && \
    __has_builtin(__builtin_amdgcn_s_wait_tensorcnt)
#define HAS_TDM 1
#else
#define HAS_TDM 0
#endif

__device__ __forceinline__ v8f wmma_f16(v16h a, v16h b, v8f c) {
    return __builtin_amdgcn_wmma_f32_16x16x32_f16(false, a, false, b, (short)0, c, false, false);
}

// --- A-fragment (16x32 f16): lane = row; halves K = k0+16*hi + j
//     (contiguous)  ->  two 16B loads.
__device__ __forceinline__ v16h load_fragA(const _Float16* base, int ld, int row0, int k0) {
    int lane = threadIdx.x & 31;
    int hi   = lane >> 4;
    const _Float16* p = base + (row0 + (lane & 15)) * ld + k0 + hi * 16;
    union { uint4 q[2]; v16h v; } t;
    t.q[0] = *(const uint4*)p;
    t.q[1] = *(const uint4*)(p + 8);
    return t.v;
}

// --- B-fragment (32x16 f16): lane = col; halves K = 16*hi + j (contiguous).
__device__ __forceinline__ v16h load_fragB_contig(const _Float16* colBase) {
    return *(const v16h*)colBase;
}

#if HAS_TDM
// LDS byte offset of a __shared__ object.
__device__ __forceinline__ unsigned int lds_offset(const void* p) {
    return (unsigned int)(size_t)((__attribute__((address_space(3))) const void*)p);
}

// Linear global->LDS DMA of `nhalf` f16 elements via the Tensor Data Mover.
// D# per CDNA5 ISA ch.8: 1-row 2D tile, data_size=2B, tile_dim0=nhalf (<=16384
// here, fits 16-bit tile field).  Call from ONE wave only; args uniform.
__device__ __forceinline__ void tdm_copy_linear(unsigned int ldsOff, const void* gptr,
                                                unsigned int nhalf) {
    unsigned long long ga = (unsigned long long)gptr;
    u32x4 g0;
    g0[0] = 1u;                                            // count=1, user D#
    g0[1] = ldsOff;                                        // lds_addr
    g0[2] = (unsigned int)(ga & 0xffffffffu);              // global_addr[31:0]
    g0[3] = (unsigned int)((ga >> 32) & 0x01ffffffu)       // global_addr[56:32]
            | 0x80000000u;                                 // type=2 ("image")
    i32x8 g1;
    g1[0] = 0x00010000;                                    // data_size=1 -> 2 bytes
    g1[1] = (int)((nhalf & 0xffffu) << 16);                // tensor_dim0[15:0]
    g1[2] = (int)((nhalf >> 16) & 0xffffu) | (1 << 16);    // tensor_dim0 hi, tensor_dim1=1
    g1[3] = (int)((nhalf & 0xffffu) << 16);                // tile_dim0
    g1[4] = 0;                                             // tile_dim1/2 unused
    g1[5] = (int)nhalf;                                    // tensor_dim0_stride[31:0]
    g1[6] = 0;
    g1[7] = 0;
    i32x4 gz4 = {0, 0, 0, 0};
    i32x8 gz8 = {0, 0, 0, 0, 0, 0, 0, 0};
    // 6-arg toolchain form: (g0, g1, g2, g3, g4, cpol)
    __builtin_amdgcn_tensor_load_to_lds(g0, g1, gz4, gz4, gz8, 0);
}
#endif

// ---------------------------------------------------------------------------
// Kernel 0a: transpose + convert x -> Xh[tok][f] f16 (LDS tiled transpose).
// ---------------------------------------------------------------------------
__global__ __launch_bounds__(256) void pack_x_kernel(const float* __restrict__ x,
                                                     _Float16* __restrict__ Xh) {
    __shared__ _Float16 Ld[F_DIM * 66];
    int tid = threadIdx.x;
    int row = blockIdx.x;                             // a*32 + bb
    const float* src = x + (long)row * (F_DIM * SEQ);
    for (int e = tid; e < F_DIM * SEQ; e += 256) {
        int f = e >> 6, t = e & 63;
        Ld[f * 66 + t] = (_Float16)src[e];
    }
    __syncthreads();
    _Float16* dst = Xh + (long)row * SEQ * F_DIM;
    for (int e = tid; e < SEQ * F_DIM; e += 256) {
        int f = e & (F_DIM - 1), t = e >> 8;
        dst[e] = Ld[f * 66 + t];
    }
}

// ---------------------------------------------------------------------------
// Kernel 0b: pack weights into B-fragment-native order:
//   Pk[(nt*8+ks)*512 + lane*16 + j] = W[k][n], n = nt*16+(lane&15),
//   k = ks*32 + 16*(lane>>4) + j.
// ---------------------------------------------------------------------------
__global__ __launch_bounds__(256) void pack_w_kernel(const float* __restrict__ Wq,
                                                     const float* __restrict__ Wk,
                                                     const float* __restrict__ Wv,
                                                     const float* __restrict__ Wp,
                                                     _Float16* __restrict__ WqPk,
                                                     _Float16* __restrict__ WkPk,
                                                     _Float16* __restrict__ WvPk,
                                                     _Float16* __restrict__ WpPk) {
    int idx    = blockIdx.x * blockDim.x + threadIdx.x;
    int stride = gridDim.x * blockDim.x;
    for (int i = idx; i < WFRAG; i += stride) {
        int j    = i & 15;
        int lane = (i >> 4) & 31;
        int tile = i >> 9;                // nt*8 + ks
        int ks   = tile & 7;
        int nt   = tile >> 3;
        int hi   = lane >> 4;
        int n    = nt * 16 + (lane & 15);
        int k    = ks * 32 + hi * 16 + j;
        int h    = n >> 5, d = n & 31;
        long sqkv = (long)h * (F_DIM * D_DIM) + (long)k * D_DIM + d;
        WqPk[i] = (_Float16)Wq[sqkv];
        WkPk[i] = (_Float16)Wk[sqkv];
        WvPk[i] = (_Float16)Wv[sqkv];
        WpPk[i] = (_Float16)Wp[(long)n * F_DIM + k];
    }
}

// ---------------------------------------------------------------------------
// Kernel 1: QKV projection (16384x256)*(256x256) x3.  Block = 128 threads
// (4 waves); block owns 64 token rows, wave owns 16.
// Output layout: Q/K/V [h][agent][b][t(64)][d(32)] f16.
// ---------------------------------------------------------------------------
__global__ __launch_bounds__(128) void qkv_kernel(const _Float16* __restrict__ Xh,
                                                  const _Float16* __restrict__ WqPk,
                                                  const _Float16* __restrict__ WkPk,
                                                  const _Float16* __restrict__ WvPk,
                                                  _Float16* __restrict__ Qh,
                                                  _Float16* __restrict__ Kh,
                                                  _Float16* __restrict__ Vh) {
    __shared__ __align__(32) _Float16 Xs[64 * F_DIM];   // 32 KB
    int tid  = threadIdx.x;
    int tok0 = blockIdx.x * 64;
#if HAS_TDM
    if (tid < 32) {
        tdm_copy_linear(lds_offset(Xs), Xh + (long)tok0 * F_DIM, 64 * F_DIM);
        __builtin_amdgcn_s_wait_tensorcnt(0);
    }
#else
    {
        const uint4* src = (const uint4*)(Xh + (long)tok0 * F_DIM);
        uint4*       dst = (uint4*)Xs;
        for (int i = tid; i < 64 * F_DIM / 8; i += 128) dst[i] = src[i];
    }
#endif
    __syncthreads();

    int wave = tid >> 5, lane = tid & 31, hi = lane >> 4;
    int row0 = wave * 16;

    v16h a[8];
#pragma unroll
    for (int ks = 0; ks < 8; ++ks) a[ks] = load_fragA(Xs, F_DIM, row0, ks * 32);

#pragma unroll
    for (int m = 0; m < 3; ++m) {
        const _Float16* WPk = (m == 0) ? WqPk : ((m == 1) ? WkPk : WvPk);
        _Float16*       O   = (m == 0) ? Qh : ((m == 1) ? Kh : Vh);
        for (int nt = 0; nt < 16; ++nt) {
            int n = nt * 16 + (lane & 15);
            int h = n >> 5, d = n & 31;
            v8f c = {};
#pragma unroll
            for (int ks = 0; ks < 8; ++ks) {
                v16h b = *(const v16h*)(WPk + ((nt * 8 + ks) * 32 + lane) * 16);
                c      = wmma_f16(a[ks], b, c);
            }
#pragma unroll
            for (int r = 0; r < 8; ++r) {
                int  tok = tok0 + row0 + r + 8 * hi;
                int  ag  = tok >> 11;
                int  bb  = (tok >> 6) & 31;
                int  t   = tok & 63;
                long oi  = ((((long)h * NAG + ag) * NB + bb) * SEQ + t) * D_DIM + d;
                O[oi]    = (_Float16)c[r];
            }
        }
    }
}

// ---------------------------------------------------------------------------
// Kernel 2: attention.  One block per (h, query-agent j, batch bb); loop over
// key-agents i: S = Q_j K_i^T * scale, softmax over s (per i, matching the
// reference), O += P V_i.  K staged by TDM (overlapped with manual V
// transpose); V staged transposed so AV B-frags are contiguous.
// ---------------------------------------------------------------------------
__global__ __launch_bounds__(128) void attn_kernel(const _Float16* __restrict__ Qh,
                                                   const _Float16* __restrict__ Kh,
                                                   const _Float16* __restrict__ Vh,
                                                   _Float16* __restrict__ Yh) {
    __shared__ __align__(32) _Float16 Qs[SEQ * D_DIM];  // [t][d]   4 KB
    __shared__ __align__(32) _Float16 Ks[SEQ * D_DIM];  // [s][d]   4 KB
    __shared__ __align__(32) _Float16 Vt[D_DIM * SEQ];  // [d][s]   4 KB (transposed)
    __shared__ __align__(32) float    Ss[SEQ * SEQ];    // 16 KB
    __shared__ __align__(32) _Float16 Ps[SEQ * SEQ];    // 8 KB

    int bid = blockIdx.x;
    int h   = bid >> 8;
    int j   = (bid >> 5) & 7;
    int bb  = bid & 31;
    int tid = threadIdx.x, wave = tid >> 5, lane = tid & 31, hi = lane >> 4;

#if HAS_TDM
    if (tid < 32) {
        tdm_copy_linear(lds_offset(Qs),
                        Qh + (((long)h * NAG + j) * NB + bb) * (SEQ * D_DIM), SEQ * D_DIM);
        __builtin_amdgcn_s_wait_tensorcnt(0);
    }
#else
    {
        const uint4* src = (const uint4*)(Qh + (((long)h * NAG + j) * NB + bb) * (SEQ * D_DIM));
        uint4*       dst = (uint4*)Qs;
        for (int e = tid; e < SEQ * D_DIM / 8; e += 128) dst[e] = src[e];
    }
#endif
    __syncthreads();

    v16h aQ = load_fragA(Qs, D_DIM, wave * 16, 0);
    v8f  o0 = {}, o1 = {};
    const float scale = 0.0625f;                        // 256^-0.5

    for (int i = 0; i < NAG; ++i) {
        __syncthreads();   // previous-iteration readers of Ks/Vt/Ps done
#if HAS_TDM
        if (tid < 32)
            tdm_copy_linear(lds_offset(Ks),
                            Kh + (((long)h * NAG + i) * NB + bb) * (SEQ * D_DIM), SEQ * D_DIM);
#else
        {
            const uint4* sk = (const uint4*)(Kh + (((long)h * NAG + i) * NB + bb) * (SEQ * D_DIM));
            uint4*       dk = (uint4*)Ks;
            for (int e = tid; e < SEQ * D_DIM / 8; e += 128) dk[e] = sk[e];
        }
#endif
        {   // V staged transposed: Vt[d][s]  (runs while the K DMA is in flight)
            const _Float16* sv = Vh + (((long)h * NAG + i) * NB + bb) * (SEQ * D_DIM);
            for (int e = tid; e < SEQ * D_DIM; e += 128) {
                int s = e >> 5, d = e & 31;
                Vt[d * SEQ + s] = sv[e];
            }
        }
#if HAS_TDM
        if (tid < 32) __builtin_amdgcn_s_wait_tensorcnt(0);
#endif
        __syncthreads();

        // scores: S[t][s] = sum_d Q[t][d]*K[s][d];  B-frag = one 32B LDS read
        for (int st = 0; st < 4; ++st) {
            int  scol = st * 16 + (lane & 15);
            v16h b    = load_fragB_contig(Ks + scol * D_DIM + hi * 16);
            v8f  c    = {};
            c         = wmma_f16(aQ, b, c);
#pragma unroll
            for (int r = 0; r < 8; ++r) {
                int trow              = wave * 16 + r + 8 * hi;
                Ss[trow * SEQ + scol] = c[r] * scale;
            }
        }
        __syncthreads();

        // softmax over s per row t (independent per key-agent i)
        if (tid < SEQ) {
            float mx = -1e30f;
            for (int s = 0; s < SEQ; ++s) mx = fmaxf(mx, Ss[tid * SEQ + s]);
            float sum = 0.f;
            for (int s = 0; s < SEQ; ++s) {
                float e = __expf(Ss[tid * SEQ + s] - mx);
                sum += e;
                Ss[tid * SEQ + s] = e;
            }
            float inv = 1.f / sum;
            for (int s = 0; s < SEQ; ++s) Ps[tid * SEQ + s] = (_Float16)(Ss[tid * SEQ + s] * inv);
        }
        __syncthreads();

        // O += P(64x64) * V(64x32); V B-frag contiguous from Vt[d][s]
#pragma unroll
        for (int ksp = 0; ksp < 2; ++ksp) {
            v16h aP = load_fragA(Ps, SEQ, wave * 16, ksp * 32);
#pragma unroll
            for (int nd = 0; nd < 2; ++nd) {
                int  dcol = nd * 16 + (lane & 15);
                v16h b    = load_fragB_contig(Vt + dcol * SEQ + ksp * 32 + hi * 16);
                if (nd == 0) o0 = wmma_f16(aP, b, o0);
                else         o1 = wmma_f16(aP, b, o1);
            }
        }
    }

    // store to Yh[tok][h*32 + d] (token-major, heads concatenated)
#pragma unroll
    for (int r = 0; r < 8; ++r) {
        int  t    = wave * 16 + r + 8 * hi;
        long tok  = ((long)j * NB + bb) * SEQ + t;
        long base = tok * F_DIM + h * D_DIM;
        Yh[base + (lane & 15)]      = (_Float16)o0[r];
        Yh[base + 16 + (lane & 15)] = (_Float16)o1[r];
    }
}

// ---------------------------------------------------------------------------
// Kernel 3: output projection  out[tok][g] = sum_f Y[tok][f]*Wp[g][f] + bp[g],
// stored transposed back to (n*b, F, SEQ) f32.
// ---------------------------------------------------------------------------
__global__ __launch_bounds__(128) void proj_kernel(const _Float16* __restrict__ Yh,
                                                   const _Float16* __restrict__ WpPk,
                                                   const float* __restrict__ bp,
                                                   float* __restrict__ out) {
    __shared__ __align__(32) _Float16 Ys[64 * F_DIM];   // 32 KB
    int tid  = threadIdx.x;
    int tok0 = blockIdx.x * 64;
#if HAS_TDM
    if (tid < 32) {
        tdm_copy_linear(lds_offset(Ys), Yh + (long)tok0 * F_DIM, 64 * F_DIM);
        __builtin_amdgcn_s_wait_tensorcnt(0);
    }
#else
    {
        const uint4* src = (const uint4*)(Yh + (long)tok0 * F_DIM);
        uint4*       dst = (uint4*)Ys;
        for (int i = tid; i < 64 * F_DIM / 8; i += 128) dst[i] = src[i];
    }
#endif
    __syncthreads();

    int wave = tid >> 5, lane = tid & 31, hi = lane >> 4;
    int row0 = wave * 16;

    v16h a[8];
#pragma unroll
    for (int ks = 0; ks < 8; ++ks) a[ks] = load_fragA(Ys, F_DIM, row0, ks * 32);

    for (int nt = 0; nt < 16; ++nt) {
        int   g    = nt * 16 + (lane & 15);
        float bias = bp[g];
        v8f   c    = {};
#pragma unroll
        for (int ks = 0; ks < 8; ++ks) {
            v16h b = *(const v16h*)(WpPk + ((nt * 8 + ks) * 32 + lane) * 16);
            c      = wmma_f16(a[ks], b, c);
        }
#pragma unroll
        for (int r = 0; r < 8; ++r) {
            int  tok = tok0 + row0 + r + 8 * hi;
            long oi  = (long)(tok >> 6) * (F_DIM * SEQ) + (long)g * SEQ + (tok & 63);
            out[oi]  = c[r] + bias;
        }
    }
}

// ---------------------------------------------------------------------------
extern "C" void kernel_launch(void* const* d_in, const int* in_sizes, int n_in,
                              void* d_out, int out_size, void* d_ws, size_t ws_size,
                              hipStream_t stream) {
    const float* x  = (const float*)d_in[0];
    const float* Wq = (const float*)d_in[1];
    const float* Wk = (const float*)d_in[2];
    const float* Wv = (const float*)d_in[3];
    const float* Wp = (const float*)d_in[4];
    const float* bp = (const float*)d_in[5];
    float*       out = (float*)d_out;

    _Float16* Xh   = (_Float16*)d_ws;                    // 16384*256
    _Float16* WqPk = Xh + (size_t)NTOK * F_DIM;
    _Float16* WkPk = WqPk + (size_t)WFRAG;
    _Float16* WvPk = WkPk + (size_t)WFRAG;
    _Float16* WpPk = WvPk + (size_t)WFRAG;
    _Float16* Qh   = WpPk + (size_t)WFRAG;
    _Float16* Kh   = Qh + (size_t)NTOK * F_DIM;
    _Float16* Vh   = Kh + (size_t)NTOK * F_DIM;
    _Float16* Yh   = Vh + (size_t)NTOK * F_DIM;

    pack_x_kernel<<<NAG * NB, 256, 0, stream>>>(x, Xh);
    pack_w_kernel<<<256, 256, 0, stream>>>(Wq, Wk, Wv, Wp, WqPk, WkPk, WvPk, WpPk);
    qkv_kernel<<<NTOK / 64, 128, 0, stream>>>(Xh, WqPk, WkPk, WvPk, Qh, Kh, Vh);
    attn_kernel<<<HEADS * NAG * NB, 128, 0, stream>>>(Qh, Kh, Vh, Yh);
    proj_kernel<<<NTOK / 64, 128, 0, stream>>>(Yh, WpPk, bp, out);
}
